// FinancialGNNRAG_67980742361614
// MI455X (gfx1250) — matrix operand order repaired and checked by reading, not compiled, not run on hardware
//
#include <hip/hip_runtime.h>
#include <hip/hip_bf16.h>

// ---------------------------------------------------------------------------
// Types for WMMA (gfx1250, wave32)
// ---------------------------------------------------------------------------
typedef __bf16 v16bf __attribute__((ext_vector_type(16)));
typedef float  v8f   __attribute__((ext_vector_type(8)));

union BfFrag {
    v16bf    v;
    unsigned u[8];
};

#define NEG_SLOPE 0.2f
#define EPS_F     1e-16f

// ---------------------------------------------------------------------------
// Helpers
// ---------------------------------------------------------------------------
__device__ __forceinline__ unsigned short f2bf(float f) {
    unsigned u = __float_as_uint(f);
    unsigned r = (u + 0x7FFFu + ((u >> 16) & 1u)) >> 16;  // round-to-nearest-even
    return (unsigned short)r;
}

// monotonic float -> uint encoding so integer atomicMax == float max
__device__ __forceinline__ unsigned fenc(float f) {
    unsigned u = __float_as_uint(f);
    return (u >> 31) ? ~u : (u | 0x80000000u);
}
__device__ __forceinline__ float fdec(unsigned u) {
    return __uint_as_float((u >> 31) ? (u & 0x7FFFFFFFu) : ~u);
}

// src/dst for edge e; e >= E are the appended self-loops
__device__ __forceinline__ void edge_sd(const int* ei, int E, int e, int& s, int& d) {
    if (e < E) { s = ei[e]; d = ei[E + e]; }
    else       { s = d = e - E; }
}

// load one 16x32 bf16 A fragment (row-major source, WMMA swizzle)
__device__ __forceinline__ void load_afrag(BfFrag& a, const unsigned short* arow,
                                           int kk, int half) {
#pragma unroll
    for (int p = 0; p < 8; ++p) {
        int j = 2 * p;
        int k = kk * 32 + j + (half << 4);                    // A: K = 16*(lane/16) + j
        a.u[p] = *(const unsigned*)(arow + k);                // (K,K+1) bf16 pair
    }
}

// ---------------------------------------------------------------------------
// Kernels
// ---------------------------------------------------------------------------

// f32 -> bf16 conversion (weights)
__global__ void cvt_bf16_k(const float* __restrict__ w, unsigned short* __restrict__ o, int n) {
    int i = blockIdx.x * blockDim.x + threadIdx.x;
    if (i < n) o[i] = f2bf(w[i]);
}

// fused ReLU (in place, f32) + bf16 copy for the next layer's WMMA A-operand
__global__ void relu_cvt_k(float* __restrict__ x, unsigned short* __restrict__ xbf,
                           long long n) {
    long long i = (long long)blockIdx.x * blockDim.x + threadIdx.x;
    if (i >= n) return;
    float v = fmaxf(x[i], 0.f);
    x[i]   = v;
    xbf[i] = f2bf(v);
}

// layer-1 linear: h[N,256] = x[N,3] @ W1[3,256]   (K=3: scalar FMA is optimal)
__global__ void lin3_k(const float* __restrict__ x, const float* __restrict__ W,
                       float* __restrict__ h, int N) {
    int i = blockIdx.x * blockDim.x + threadIdx.x;
    if (i >= N * 256) return;
    int n = i >> 8, c = i & 255;
    const float* xr = x + (size_t)n * 3;
    h[i] = fmaf(xr[0], W[c], fmaf(xr[1], W[256 + c], xr[2] * W[512 + c]));
}

// bf16 WMMA GEMM: C[M,NC] = Abf[M,256] (bf16) @ Wbf[256,NC] (bf16), f32 accum
// block = 1 wave (32 threads); grid = (ceil(M/16 / RPW), NC/16)
// A fragments double-buffered so loads overlap the WMMA pipeline.
#define GEMM_RPW 4
__global__ void wmma_gemm_k(const unsigned short* __restrict__ Abf,
                            const unsigned short* __restrict__ Wbf,
                            float* __restrict__ C, int M, int NC) {
    const int lane = threadIdx.x;        // 0..31
    const int half = lane >> 4;          // 0/1
    const int l15  = lane & 15;
    const int col  = blockIdx.y * 16 + l15;

    // preload all B fragments for this column tile (K = 8 steps of 32)
    BfFrag b[8];
#pragma unroll
    for (int kk = 0; kk < 8; ++kk) {
#pragma unroll
        for (int p = 0; p < 8; ++p) {
            int kb = kk * 32 + (half << 4) + 2 * p;   // B: K = 16*(lane/16) + j
            unsigned lo = Wbf[(size_t)kb * NC + col];
            unsigned hi = Wbf[(size_t)(kb + 1) * NC + col];
            b[kk].u[p] = lo | (hi << 16);
        }
    }

    const int tiles = M >> 4;
    for (int r = 0; r < GEMM_RPW; ++r) {
        int tile = blockIdx.x * GEMM_RPW + r;
        if (tile >= tiles) break;
        int row = tile * 16 + l15;                    // A: row M = lane%16
        const unsigned short* arow = Abf + (size_t)row * 256;

        v8f c = {0.f, 0.f, 0.f, 0.f, 0.f, 0.f, 0.f, 0.f};
        BfFrag a[2];
        load_afrag(a[0], arow, 0, half);
#pragma unroll
        for (int kk = 0; kk < 8; ++kk) {
            if (kk + 1 < 8)                            // prefetch next fragment
                load_afrag(a[(kk + 1) & 1], arow, kk + 1, half);
            c = __builtin_amdgcn_wmma_f32_16x16x32_bf16(
                    false, a[kk & 1].v, false, b[kk].v, (short)0, c, false, false);
        }

        float* cbase = C + (size_t)tile * 16 * NC + col;
#pragma unroll
        for (int v = 0; v < 8; ++v)                   // C: row = v + 8*(lane/16)
            cbase[(size_t)(v + 8 * half) * NC] = c[v];
    }
}

// per-node attention logits: as[n,h] = <h[n,h,:], a_src[h,:]>, same for ad
__global__ void att_k(const float* __restrict__ hbuf,
                      const float* __restrict__ asrc, const float* __restrict__ adst,
                      float* __restrict__ as_, float* __restrict__ ad_,
                      int N, int H, int C) {
    int i = blockIdx.x * blockDim.x + threadIdx.x;
    if (i >= N * H) return;
    int n = i / H, hd = i - n * H;
    const float* hp = hbuf + (size_t)n * H * C + (size_t)hd * C;
    const float* pa = asrc + (size_t)hd * C;
    const float* pd = adst + (size_t)hd * C;
    float sa = 0.f, sd = 0.f;
    for (int c = 0; c < C; ++c) {
        float v = hp[c];
        sa = fmaf(v, pa[c], sa);
        sd = fmaf(v, pd[c], sd);
    }
    as_[i] = sa;
    ad_[i] = sd;
}

// init output accumulator with bias broadcast
__global__ void init_out_k(float* __restrict__ out, const float* __restrict__ b,
                           long long total, int HC) {
    long long i = (long long)blockIdx.x * blockDim.x + threadIdx.x;
    if (i < total) out[i] = b[(int)(i % HC)];
}

// init segment-max (encoded) and segment-sum
__global__ void init_ms_k(unsigned* __restrict__ m, float* __restrict__ s, int n) {
    int i = blockIdx.x * blockDim.x + threadIdx.x;
    if (i < n) { m[i] = 0u; s[i] = 0.f; }
}

// edge pass 1: alpha = leaky_relu(as[src]+ad[dst]); store; segment max
__global__ void edge_p1_k(const int* __restrict__ ei, int E,
                          const float* __restrict__ as_, const float* __restrict__ ad_,
                          float* __restrict__ ebuf, unsigned* __restrict__ m,
                          int H, int ET) {
    int tid = blockIdx.x * blockDim.x + threadIdx.x;
    if (tid >= ET * H) return;
    int e = tid / H, hd = tid - e * H;
    int s, d;
    edge_sd(ei, E, e, s, d);
    float a = as_[s * H + hd] + ad_[d * H + hd];
    a = (a > 0.f) ? a : a * NEG_SLOPE;
    ebuf[tid] = a;
    atomicMax(m + d * H + hd, fenc(a));
}

// edge pass 2: e = exp(alpha - m[dst]); store; segment sum
__global__ void edge_p2_k(const int* __restrict__ ei, int E,
                          float* __restrict__ ebuf, const unsigned* __restrict__ m,
                          float* __restrict__ sbuf, int H, int ET) {
    int tid = blockIdx.x * blockDim.x + threadIdx.x;
    if (tid >= ET * H) return;
    int e = tid / H, hd = tid - e * H;
    int s, d;
    edge_sd(ei, E, e, s, d);
    float a  = ebuf[tid];
    float ev = __expf(a - fdec(m[d * H + hd]));
    ebuf[tid] = ev;
    atomicAdd(sbuf + d * H + hd, ev);
}

// edge pass 3: out[dst] += h[src] * (e / (s[dst]+eps)); float4 gather, f32 atomics
__global__ void edge_aggr_k(const int* __restrict__ ei, int E,
                            const float* __restrict__ hin, const float* __restrict__ ebuf,
                            const float* __restrict__ sbuf, float* __restrict__ out,
                            int H, int C, int ET) {
    const int chunks = (H * C) >> 2;
    long long tid = (long long)blockIdx.x * blockDim.x + threadIdx.x;
    if (tid >= (long long)ET * chunks) return;
    int e  = (int)(tid / chunks);
    int c4 = ((int)(tid - (long long)e * chunks)) << 2;
    int hd = c4 / C;
    int s, d;
    edge_sd(ei, E, e, s, d);
    float w = ebuf[(size_t)e * H + hd] / (sbuf[(size_t)d * H + hd] + EPS_F);
    const float4 v = *(const float4*)(hin + (size_t)s * H * C + c4);
    float* o = out + (size_t)d * H * C + c4;
    atomicAdd(o + 0, v.x * w);
    atomicAdd(o + 1, v.y * w);
    atomicAdd(o + 2, v.z * w);
    atomicAdd(o + 3, v.w * w);
}

// ---------------------------------------------------------------------------
// Launch
// ---------------------------------------------------------------------------
static inline int nblk(long long n, int b) { return (int)((n + b - 1) / b); }

extern "C" void kernel_launch(void* const* d_in, const int* in_sizes, int n_in,
                              void* d_out, int out_size, void* d_ws, size_t ws_size,
                              hipStream_t stream) {
    const float* x       = (const float*)d_in[0];
    const int*   ei      = (const int*)d_in[1];
    const float* W1      = (const float*)d_in[2];
    const float* a_src1  = (const float*)d_in[3];
    const float* a_dst1  = (const float*)d_in[4];
    const float* b1      = (const float*)d_in[5];
    const float* W2      = (const float*)d_in[6];
    const float* a_src2  = (const float*)d_in[7];
    const float* a_dst2  = (const float*)d_in[8];
    const float* b2      = (const float*)d_in[9];
    const float* W3      = (const float*)d_in[10];
    const float* a_src3  = (const float*)d_in[11];
    const float* a_dst3  = (const float*)d_in[12];
    const float* b3      = (const float*)d_in[13];
    float*       outp    = (float*)d_out;

    const int N  = in_sizes[0] / 3;
    const int E  = in_sizes[1] / 2;
    const int ET = E + N;

    // workspace carve-up
    char* w = (char*)d_ws;
    float* hA = (float*)w;            w += (size_t)N * 256 * 4;   // linear outputs h
    float* hB = (float*)w;            w += (size_t)N * 256 * 4;   // aggregation outputs
    unsigned short* hBbf = (unsigned short*)w; w += (size_t)N * 256 * 2; // bf16 copy of hB
    unsigned short* wbf  = (unsigned short*)w; w += (size_t)256 * 256 * 2;
    float*    as_  = (float*)w;       w += (size_t)N * 4 * 4;
    float*    ad_  = (float*)w;       w += (size_t)N * 4 * 4;
    unsigned* mbuf = (unsigned*)w;    w += (size_t)N * 4 * 4;
    float*    sbuf = (float*)w;       w += (size_t)N * 4 * 4;
    float*    ebuf = (float*)w;       w += (size_t)ET * 4 * 4;

    const int B = 256;
    const int rowTiles = N / 16;                       // N = 100000 = 16*6250
    dim3 g256(nblk((long long)rowTiles, GEMM_RPW), 16);
    dim3 g32 (nblk((long long)rowTiles, GEMM_RPW), 2);

    // ---------------- layer 1 (in=3 -> H=4, C=64, concat, relu) --------------
    lin3_k   <<<nblk((long long)N * 256, B), B, 0, stream>>>(x, W1, hA, N);
    att_k    <<<nblk((long long)N * 4, B),  B, 0, stream>>>(hA, a_src1, a_dst1, as_, ad_, N, 4, 64);
    init_out_k<<<nblk((long long)N * 256, B), B, 0, stream>>>(hB, b1, (long long)N * 256, 256);
    init_ms_k<<<nblk((long long)N * 4, B),  B, 0, stream>>>(mbuf, sbuf, N * 4);
    edge_p1_k<<<nblk((long long)ET * 4, B), B, 0, stream>>>(ei, E, as_, ad_, ebuf, mbuf, 4, ET);
    edge_p2_k<<<nblk((long long)ET * 4, B), B, 0, stream>>>(ei, E, ebuf, mbuf, sbuf, 4, ET);
    edge_aggr_k<<<nblk((long long)ET * 64, B), B, 0, stream>>>(ei, E, hA, ebuf, sbuf, hB, 4, 64, ET);
    relu_cvt_k<<<nblk((long long)N * 256, B), B, 0, stream>>>(hB, hBbf, (long long)N * 256);

    // ---------------- layer 2 (256 -> H=4, C=64, concat, relu) ---------------
    cvt_bf16_k<<<nblk(256 * 256, B), B, 0, stream>>>(W2, wbf, 256 * 256);
    wmma_gemm_k<<<g256, 32, 0, stream>>>(hBbf, wbf, hA, N, 256);     // hA = relu(hB) @ W2
    att_k    <<<nblk((long long)N * 4, B),  B, 0, stream>>>(hA, a_src2, a_dst2, as_, ad_, N, 4, 64);
    init_out_k<<<nblk((long long)N * 256, B), B, 0, stream>>>(hB, b2, (long long)N * 256, 256);
    init_ms_k<<<nblk((long long)N * 4, B),  B, 0, stream>>>(mbuf, sbuf, N * 4);
    edge_p1_k<<<nblk((long long)ET * 4, B), B, 0, stream>>>(ei, E, as_, ad_, ebuf, mbuf, 4, ET);
    edge_p2_k<<<nblk((long long)ET * 4, B), B, 0, stream>>>(ei, E, ebuf, mbuf, sbuf, 4, ET);
    edge_aggr_k<<<nblk((long long)ET * 64, B), B, 0, stream>>>(ei, E, hA, ebuf, sbuf, hB, 4, 64, ET);
    relu_cvt_k<<<nblk((long long)N * 256, B), B, 0, stream>>>(hB, hBbf, (long long)N * 256);

    // ---------------- layer 3 (256 -> H=1, C=32, mean==identity) -------------
    cvt_bf16_k<<<nblk(256 * 32, B), B, 0, stream>>>(W3, wbf, 256 * 32);
    wmma_gemm_k<<<g32, 32, 0, stream>>>(hBbf, wbf, hA, N, 32);       // hA = relu(hB) @ W3
    att_k    <<<nblk((long long)N, B),      B, 0, stream>>>(hA, a_src3, a_dst3, as_, ad_, N, 1, 32);
    init_out_k<<<nblk((long long)N * 32, B), B, 0, stream>>>(outp, b3, (long long)N * 32, 32);
    init_ms_k<<<nblk((long long)N, B),      B, 0, stream>>>(mbuf, sbuf, N);
    edge_p1_k<<<nblk((long long)ET, B),     B, 0, stream>>>(ei, E, as_, ad_, ebuf, mbuf, 1, ET);
    edge_p2_k<<<nblk((long long)ET, B),     B, 0, stream>>>(ei, E, ebuf, mbuf, sbuf, 1, ET);
    edge_aggr_k<<<nblk((long long)ET * 8, B), B, 0, stream>>>(ei, E, hA, ebuf, sbuf, outp, 1, 32, ET);
}
